// Mamba2Layer_37580963840655
// MI455X (gfx1250) — matrix operand order repaired and checked by reading, not compiled, not run on hardware
//
#include <hip/hip_runtime.h>

// ---- problem constants ----
#define B_    2
#define L_    4096
#define DM    1024
#define DI    2048
#define NH    32
#define HD    64
#define DST   128
#define CONVD 2304
#define NPROJ 4384
#define NPAD  4480      // 4384 padded to multiple of 128 for clean WMMA block tiling
#define NC    32        // chunks
#define CK    128       // chunk length

typedef __attribute__((ext_vector_type(16))) __bf16 v16bf;
typedef __attribute__((ext_vector_type(8)))  __bf16 v8bf;
typedef __attribute__((ext_vector_type(8)))  float  v8f;

static __device__ __forceinline__ __bf16 f2bf(float f) {
  unsigned u = __builtin_bit_cast(unsigned, f);
  u += 0x7fffu + ((u >> 16) & 1u);            // round-to-nearest-even
  unsigned short h = (unsigned short)(u >> 16);
  return __builtin_bit_cast(__bf16, h);
}
static __device__ __forceinline__ float bf2f(__bf16 b) {
  unsigned u = ((unsigned)__builtin_bit_cast(unsigned short, b)) << 16;
  return __builtin_bit_cast(float, u);
}

// A fragment: 16 rows x 32 K from row-major src (ld elements), base at (row0,k0).
// Lane 0-15: M=lane, K {0..7, 16..23}; lane 16-31: M=lane-16, K {8..15, 24..31}.
static __device__ __forceinline__ v16bf fragA(const __bf16* src, int ld, int lane) {
  int r  = lane & 15;
  int hf = lane >> 4;
  const __bf16* p = src + r * ld + hf * 8;
  v8bf lo = *(const v8bf*)p;
  v8bf hi = *(const v8bf*)(p + 16);
  return __builtin_shufflevector(lo, hi, 0,1,2,3,4,5,6,7,8,9,10,11,12,13,14,15);
}
// B fragment loaded from row-major B^T (rows = N, cols = K).
// Lane 0-15: N=lane, K {0..7, 16..23}; lane 16-31: N=lane-16, K {8..15, 24..31}.
static __device__ __forceinline__ v16bf fragB(const __bf16* srcT, int ld, int lane) {
  int n  = lane & 15;
  int hf = lane >> 4;
  const __bf16* p = srcT + n * ld + hf * 8;
  v8bf lo = *(const v8bf*)p;
  v8bf hi = *(const v8bf*)(p + 16);
  return __builtin_shufflevector(lo, hi, 0,1,2,3,4,5,6,7,8,9,10,11,12,13,14,15);
}
static __device__ __forceinline__ v8f wmma_bf16(v16bf a, v16bf b, v8f c) {
  return __builtin_amdgcn_wmma_f32_16x16x32_bf16(false, a, false, b, (short)0, c, false, false);
}
static __device__ __forceinline__ void zero8(v8f* a, int n) {
  for (int t = 0; t < n; ++t)
    for (int i = 0; i < 8; ++i) a[t][i] = 0.f;
}

// ---------------- layernorm -> bf16 activations ----------------
__global__ __launch_bounds__(256) void k_ln(const float* __restrict__ x,
                                            const float* __restrict__ w,
                                            const float* __restrict__ bsh,
                                            __bf16* __restrict__ xnbf) {
  __shared__ float red[256];
  int row = blockIdx.x, tid = threadIdx.x;
  const float* xr = x + (size_t)row * DM;
  float v[4];
  float s = 0.f;
  for (int j = 0; j < 4; ++j) { float t = xr[tid + 256 * j]; v[j] = t; s += t; }
  red[tid] = s; __syncthreads();
  for (int st = 128; st > 0; st >>= 1) { if (tid < st) red[tid] += red[tid + st]; __syncthreads(); }
  float mu = red[0] * (1.f / DM);
  __syncthreads();
  float s2 = 0.f;
  for (int j = 0; j < 4; ++j) { float d = v[j] - mu; s2 += d * d; }
  red[tid] = s2; __syncthreads();
  for (int st = 128; st > 0; st >>= 1) { if (tid < st) red[tid] += red[tid + st]; __syncthreads(); }
  float rs = rsqrtf(red[0] * (1.f / DM) + 1e-5f);
  for (int j = 0; j < 4; ++j) {
    int i = tid + 256 * j;
    xnbf[(size_t)row * DM + i] = f2bf((v[j] - mu) * rs * w[i] + bsh[i]);
  }
}

// ---------------- weight conversions ----------------
__global__ void k_cvt_w1(const float* __restrict__ w, __bf16* __restrict__ o) {
  int i = blockIdx.x * 256 + threadIdx.x;        // over NPAD*DM
  int n = i / DM;
  o[i] = (n < NPROJ) ? f2bf(w[i]) : f2bf(0.f);
}
__global__ void k_cvt_w2(const float* __restrict__ w, __bf16* __restrict__ o) {
  int i = blockIdx.x * 256 + threadIdx.x;        // over DM*DI
  o[i] = f2bf(w[i]);
}

// ---------------- generic bf16 WMMA GEMM: C[M,N] = A[M,K] * Bt[N,K]^T (+resid) ----------------
// Block: 256 threads = 8 waves as 4(M) x 2(N); block tile 128x128, wave tile 32x64.
// Double-buffered fragment prefetch so global_load_b128 overlaps the 8-WMMA burst.
__global__ __launch_bounds__(256) void k_gemm(const __bf16* __restrict__ A, int lda,
                                              const __bf16* __restrict__ Bt, int ldb,
                                              float* __restrict__ C, int ldc,
                                              const float* __restrict__ resid, int ldr,
                                              int K) {
  int tid = threadIdx.x, lane = tid & 31, w = tid >> 5;
  int wm = w & 3, wn = w >> 2;
  int m0 = blockIdx.y * 128 + wm * 32;
  int n0 = blockIdx.x * 128 + wn * 64;
  const __bf16* Ap = A + (size_t)m0 * lda;
  const __bf16* Bp = Bt + (size_t)n0 * ldb;

  v8f acc[8]; zero8(acc, 8);
  v16bf a0 = fragA(Ap, lda, lane);
  v16bf a1 = fragA(Ap + 16 * lda, lda, lane);
  v16bf b0 = fragB(Bp, ldb, lane);
  v16bf b1 = fragB(Bp + 16 * ldb, ldb, lane);
  v16bf b2 = fragB(Bp + 32 * ldb, ldb, lane);
  v16bf b3 = fragB(Bp + 48 * ldb, ldb, lane);

  for (int k0 = 0; k0 < K; k0 += 32) {
    int k1 = k0 + 32;
    int kn = (k1 < K) ? k1 : k0;               // branch-free tail: harmless reload
    v16bf na0 = fragA(Ap + kn, lda, lane);
    v16bf na1 = fragA(Ap + 16 * lda + kn, lda, lane);
    v16bf nb0 = fragB(Bp + kn, ldb, lane);
    v16bf nb1 = fragB(Bp + 16 * ldb + kn, ldb, lane);
    v16bf nb2 = fragB(Bp + 32 * ldb + kn, ldb, lane);
    v16bf nb3 = fragB(Bp + 48 * ldb + kn, ldb, lane);

    acc[0] = wmma_bf16(a0, b0, acc[0]);
    acc[1] = wmma_bf16(a0, b1, acc[1]);
    acc[2] = wmma_bf16(a0, b2, acc[2]);
    acc[3] = wmma_bf16(a0, b3, acc[3]);
    acc[4] = wmma_bf16(a1, b0, acc[4]);
    acc[5] = wmma_bf16(a1, b1, acc[5]);
    acc[6] = wmma_bf16(a1, b2, acc[6]);
    acc[7] = wmma_bf16(a1, b3, acc[7]);

    a0 = na0; a1 = na1; b0 = nb0; b1 = nb1; b2 = nb2; b3 = nb3;
  }

  int cl = lane & 15, hf = lane >> 4;
  for (int tm = 0; tm < 2; ++tm)
    for (int tn = 0; tn < 4; ++tn)
      for (int r = 0; r < 8; ++r) {
        int row = m0 + tm * 16 + r + 8 * hf;
        int col = n0 + tn * 16 + cl;
        float vv = acc[tm * 4 + tn][r];
        if (resid) vv += resid[(size_t)row * ldr + col];
        C[(size_t)row * ldc + col] = vv;
      }
}

// ---------------- dt = softplus(raw + bias), a = dt * (-exp(A_log)) ----------------
__global__ void k_dt(const float* __restrict__ zx, const float* __restrict__ dt_bias,
                     const float* __restrict__ A_log, float* __restrict__ dt,
                     float* __restrict__ a) {
  int i = blockIdx.x * 256 + threadIdx.x;        // B_*L_*NH
  int h = i & 31, row = i >> 5;
  float v = zx[(size_t)row * NPAD + (DI + CONVD) + h] + dt_bias[h];
  float sp = (v > 20.f) ? v : log1pf(__expf(v));
  dt[i] = sp;
  a[i] = -sp * __expf(A_log[h]);
}

// ---------------- depthwise causal conv (width 4) + SiLU ----------------
__global__ void k_conv(const float* __restrict__ zx, const float* __restrict__ cw,
                       const float* __restrict__ cb, float* __restrict__ xbc) {
  int i = blockIdx.x * 256 + threadIdx.x;        // B_*L_*CONVD
  int ch = i % CONVD;
  int bl = i / CONVD;
  int l = bl % L_, b = bl / L_;
  float acc = cb[ch];
#pragma unroll
  for (int j = 0; j < 4; ++j) {
    int ls = l - 3 + j;
    if (ls >= 0) acc += cw[ch * 4 + j] * zx[((size_t)(b * L_ + ls)) * NPAD + DI + ch];
  }
  xbc[i] = acc / (1.f + __expf(-acc));
}

// ---------------- G = Cc @ Bc^T per (b, chunk), bf16 output ----------------
__global__ __launch_bounds__(256) void k_gmat(const float* __restrict__ xbc,
                                              __bf16* __restrict__ gbf) {
  __shared__ __align__(16) __bf16 Cs[CK * DST];
  __shared__ __align__(16) __bf16 Bs[CK * DST];
  int blk = blockIdx.x;
  int c = blk & 31, b = blk >> 5;
  int tid = threadIdx.x, lane = tid & 31, w = tid >> 5;
  for (int i = tid; i < CK * DST; i += 256) {
    int l = i >> 7, n = i & 127;
    size_t base = ((size_t)(b * L_ + c * CK + l)) * CONVD;
    Bs[i] = f2bf(xbc[base + DI + n]);
    Cs[i] = f2bf(xbc[base + DI + DST + n]);
  }
  __syncthreads();
  v8f acc[8]; zero8(acc, 8);
  for (int k0 = 0; k0 < DST; k0 += 32) {
    v16bf af = fragA(Cs + w * 16 * DST + k0, DST, lane);
#pragma unroll
    for (int t = 0; t < 8; ++t) {
      v16bf bf = fragB(Bs + 16 * t * DST + k0, DST, lane);
      acc[t] = wmma_bf16(af, bf, acc[t]);
    }
  }
  int cl = lane & 15, hf = lane >> 4;
  __bf16* g = gbf + (size_t)blk * (CK * CK);
  for (int t = 0; t < 8; ++t)
    for (int r = 0; r < 8; ++r) {
      int l = w * 16 + r + 8 * hf, s = 16 * t + cl;
      g[l * CK + s] = f2bf(acc[t][r]);
    }
}

// ---------------- per (b,c,h): Y_diag and chunk states ----------------
__global__ __launch_bounds__(256) void k_s1(const float* __restrict__ xbc,
                                            const float* __restrict__ a_g,
                                            const float* __restrict__ dt_g,
                                            const __bf16* __restrict__ gbf,
                                            float* __restrict__ y,
                                            float* __restrict__ acum_g,
                                            float* __restrict__ asum_g,
                                            float* __restrict__ states_g) {
  __shared__ float ac[CK];
  __shared__ __align__(16) __bf16 Ms[CK * CK];   // first L-masked matrix, then reused as Bd^T
  __shared__ __align__(16) __bf16 XT[HD * CK];   // X^T [p][l]
  int blk = blockIdx.x;
  int h = blk & 31, c = (blk >> 5) & 31, b = blk >> 10;
  int tid = threadIdx.x, lane = tid & 31, w = tid >> 5;

  if (tid < CK) ac[tid] = a_g[((size_t)(b * L_ + c * CK + tid)) * NH + h];
  __syncthreads();
  if (tid == 0) { float s = 0.f; for (int l = 0; l < CK; ++l) { s += ac[l]; ac[l] = s; } }
  __syncthreads();
  float asum = ac[CK - 1];
  if (tid < CK) acum_g[((size_t)(b * NH + h)) * L_ + c * CK + tid] = ac[tid];
  if (tid == 0) asum_g[(b * NH + h) * NC + c] = asum;

  const __bf16* g = gbf + ((size_t)(b * NC + c)) * (CK * CK);
  for (int i = tid; i < CK * CK; i += 256) {
    int l = i >> 7, s = i & 127;
    float v = (s <= l) ? bf2f(g[i]) * __expf(ac[l] - ac[s]) : 0.f;
    Ms[i] = f2bf(v);
  }
  for (int i = tid; i < HD * CK; i += 256) {
    int p = i >> 7, l = i & 127;
    size_t lg = (size_t)(b * L_ + c * CK + l);
    XT[i] = f2bf(xbc[lg * CONVD + h * HD + p] * dt_g[lg * NH + h]);
  }
  __syncthreads();

  // Y_diag = (G .* Lmat) @ X : [128 x 128] @ [128 x 64]
  v8f acc[4]; zero8(acc, 4);
  for (int k0 = 0; k0 < CK; k0 += 32) {
    v16bf af = fragA(Ms + w * 16 * CK + k0, CK, lane);
#pragma unroll
    for (int t = 0; t < 4; ++t) {
      v16bf bf = fragB(XT + 16 * t * CK + k0, CK, lane);
      acc[t] = wmma_bf16(af, bf, acc[t]);
    }
  }
  int cl = lane & 15, hf = lane >> 4;
  for (int t = 0; t < 4; ++t)
    for (int r = 0; r < 8; ++r) {
      int l = w * 16 + r + 8 * hf, p = 16 * t + cl;
      y[((size_t)(b * L_ + c * CK + l)) * DI + h * HD + p] = acc[t][r];
    }
  __syncthreads();                                // done reading Ms

  // rebuild Ms as Bd^T[n][l] = Bc[l,n] * exp(asum - acum[l])
  for (int i = tid; i < CK * CK; i += 256) {
    int n = i >> 7, l = i & 127;
    size_t lg = (size_t)(b * L_ + c * CK + l);
    Ms[i] = f2bf(xbc[lg * CONVD + DI + n] * __expf(asum - ac[l]));
  }
  __syncthreads();

  // states[n,p] = Bd^T @ X : [128 x 128] @ [128 x 64]
  v8f acc2[4]; zero8(acc2, 4);
  for (int k0 = 0; k0 < CK; k0 += 32) {
    v16bf af = fragA(Ms + w * 16 * CK + k0, CK, lane);
#pragma unroll
    for (int t = 0; t < 4; ++t) {
      v16bf bf = fragB(XT + 16 * t * CK + k0, CK, lane);
      acc2[t] = wmma_bf16(af, bf, acc2[t]);
    }
  }
  float* st = states_g + (size_t)blk * (DST * HD);
  for (int t = 0; t < 4; ++t)
    for (int r = 0; r < 8; ++r) {
      int n = w * 16 + r + 8 * hf, p = 16 * t + cl;
      st[n * HD + p] = acc2[t][r];
    }
}

// ---------------- sequential inter-chunk scan per (b,h); emits prev^T in bf16 ----------------
__global__ __launch_bounds__(256) void k_s2(const float* __restrict__ states_g,
                                            const float* __restrict__ asum_g,
                                            __bf16* __restrict__ prevT) {
  int blk = blockIdx.x;
  int h = blk & 31, b = blk >> 5;
  int tid = threadIdx.x;
  float S[32];
  for (int j = 0; j < 32; ++j) S[j] = 0.f;
  for (int c = 0; c < NC; ++c) {
    float dec = __expf(asum_g[(b * NH + h) * NC + c]);
    size_t base = ((size_t)((b * NC + c) * NH + h)) * (DST * HD);
    for (int j = 0; j < 32; ++j) {
      int id = tid + 256 * j;                 // id = n*64 + p
      int n = id >> 6, p = id & 63;
      prevT[base + p * DST + n] = f2bf(S[j]); // state entering chunk c, transposed
      S[j] = S[j] * dec + states_g[base + id];
    }
  }
}

// ---------------- per (b,c,h): Y_off = (exp(acum).*C) @ prev, fold +D*xs ----------------
__global__ __launch_bounds__(256) void k_s3(const float* __restrict__ xbc,
                                            const float* __restrict__ acum_g,
                                            const __bf16* __restrict__ prevT,
                                            const float* __restrict__ Dp,
                                            float* __restrict__ y) {
  __shared__ float ac[CK];
  __shared__ __align__(16) __bf16 Ce[CK * DST];
  int blk = blockIdx.x;
  int h = blk & 31, c = (blk >> 5) & 31, b = blk >> 10;
  int tid = threadIdx.x, lane = tid & 31, w = tid >> 5;
  if (tid < CK) ac[tid] = acum_g[((size_t)(b * NH + h)) * L_ + c * CK + tid];
  __syncthreads();
  for (int i = tid; i < CK * DST; i += 256) {
    int l = i >> 7, n = i & 127;
    size_t lg = (size_t)(b * L_ + c * CK + l);
    Ce[i] = f2bf(xbc[lg * CONVD + DI + DST + n] * __expf(ac[l]));
  }
  __syncthreads();
  const __bf16* pv = prevT + (size_t)blk * (DST * HD);   // [p][n], ld = 128
  v8f acc[4]; zero8(acc, 4);
  for (int k0 = 0; k0 < DST; k0 += 32) {
    v16bf af = fragA(Ce + w * 16 * DST + k0, DST, lane);
#pragma unroll
    for (int t = 0; t < 4; ++t) {
      v16bf bf = fragB(pv + 16 * t * DST + k0, DST, lane);
      acc[t] = wmma_bf16(af, bf, acc[t]);
    }
  }
  int cl = lane & 15, hf = lane >> 4;
  float dh = Dp[h];
  for (int t = 0; t < 4; ++t)
    for (int r = 0; r < 8; ++r) {
      int l = w * 16 + r + 8 * hf, p = 16 * t + cl;
      size_t lg = (size_t)(b * L_ + c * CK + l);
      size_t yi = lg * DI + h * HD + p;
      y[yi] = y[yi] + acc[t][r] + dh * xbc[lg * CONVD + h * HD + p];
    }
}

// ---------------- gating (SiLU(z)) + RMSNorm, emit bf16 for out_proj ----------------
__global__ __launch_bounds__(256) void k_gate(const float* __restrict__ y,
                                              const float* __restrict__ zx,
                                              const float* __restrict__ rms_w,
                                              __bf16* __restrict__ ybf) {
  __shared__ float red[256];
  int row = blockIdx.x, tid = threadIdx.x;
  float v[8];
  float ss = 0.f;
  for (int j = 0; j < 8; ++j) {
    int i = tid + 256 * j;
    float yv = y[(size_t)row * DI + i];
    float zv = zx[(size_t)row * NPAD + i];
    yv *= zv / (1.f + __expf(-zv));
    v[j] = yv; ss += yv * yv;
  }
  red[tid] = ss; __syncthreads();
  for (int st = 128; st > 0; st >>= 1) { if (tid < st) red[tid] += red[tid + st]; __syncthreads(); }
  float sc = rsqrtf(red[0] * (1.f / DI) + 1e-5f);
  for (int j = 0; j < 8; ++j) {
    int i = tid + 256 * j;
    ybf[(size_t)row * DI + i] = f2bf(v[j] * sc * rms_w[i]);
  }
}

extern "C" void kernel_launch(void* const* d_in, const int* in_sizes, int n_in,
                              void* d_out, int out_size, void* d_ws, size_t ws_size,
                              hipStream_t stream) {
  (void)in_sizes; (void)n_in; (void)out_size; (void)ws_size;
  const float* x        = (const float*)d_in[0];
  const float* norm_w   = (const float*)d_in[1];
  const float* norm_b   = (const float*)d_in[2];
  const float* in_proj  = (const float*)d_in[3];
  const float* conv_w   = (const float*)d_in[4];
  const float* conv_b   = (const float*)d_in[5];
  const float* dt_bias  = (const float*)d_in[6];
  const float* A_log    = (const float*)d_in[7];
  const float* Dp       = (const float*)d_in[8];
  const float* rms_w    = (const float*)d_in[9];
  const float* out_proj = (const float*)d_in[10];
  float* out = (float*)d_out;

  char* p = (char*)d_ws;
  auto alloc = [&](size_t bytes) -> void* {
    void* r = (void*)p;
    p += (bytes + 255) & ~(size_t)255;
    return r;
  };
  const size_t BL = (size_t)B_ * L_;
  __bf16* xnbf  = (__bf16*)alloc(BL * DM * 2);
  __bf16* w1bf  = (__bf16*)alloc((size_t)NPAD * DM * 2);
  __bf16* w2bf  = (__bf16*)alloc((size_t)DM * DI * 2);
  float*  zx    = (float*) alloc(BL * NPAD * 4);
  float*  xbc   = (float*) alloc(BL * CONVD * 4);
  float*  dt    = (float*) alloc(BL * NH * 4);
  float*  a     = (float*) alloc(BL * NH * 4);
  float*  acum  = (float*) alloc(BL * NH * 4);
  float*  asum  = (float*) alloc((size_t)B_ * NH * NC * 4);
  __bf16* gbf   = (__bf16*)alloc((size_t)B_ * NC * CK * CK * 2);
  float*  states= (float*) alloc((size_t)B_ * NC * NH * DST * HD * 4);
  __bf16* prevT = (__bf16*)alloc((size_t)B_ * NC * NH * DST * HD * 2);
  float*  y     = (float*) alloc(BL * DI * 4);
  __bf16* ybf   = (__bf16*)alloc(BL * DI * 2);

  k_ln<<<(unsigned)BL, 256, 0, stream>>>(x, norm_w, norm_b, xnbf);
  k_cvt_w1<<<(NPAD * DM) / 256, 256, 0, stream>>>(in_proj, w1bf);
  k_cvt_w2<<<(DM * DI) / 256, 256, 0, stream>>>(out_proj, w2bf);
  k_gemm<<<dim3(NPAD / 128, (unsigned)(BL / 128)), 256, 0, stream>>>(
      xnbf, DM, w1bf, DM, zx, NPAD, nullptr, 0, DM);
  k_dt<<<(unsigned)((BL * NH) / 256), 256, 0, stream>>>(zx, dt_bias, A_log, dt, a);
  k_conv<<<(unsigned)((BL * CONVD) / 256), 256, 0, stream>>>(zx, conv_w, conv_b, xbc);
  k_gmat<<<B_ * NC, 256, 0, stream>>>(xbc, gbf);
  k_s1<<<B_ * NC * NH, 256, 0, stream>>>(xbc, a, dt, gbf, y, acum, asum, states);
  k_s2<<<B_ * NH, 256, 0, stream>>>(states, asum, prevT);
  k_s3<<<B_ * NC * NH, 256, 0, stream>>>(xbc, acum, prevT, Dp, y);
  k_gate<<<(unsigned)BL, 256, 0, stream>>>(y, zx, rms_w, ybf);
  k_gemm<<<dim3(DM / 128, (unsigned)(BL / 128)), 256, 0, stream>>>(
      ybf, DI, w2bf, DI, out, DM, x, DM, DI);
}
